// SAM_3616362463796
// MI455X (gfx1250) — matrix operand order repaired and checked by reading, not compiled, not run on hardware
//
#include <hip/hip_runtime.h>
#include <hip/hip_bf16.h>

typedef __attribute__((ext_vector_type(16))) _Float16 v16h;
typedef __attribute__((ext_vector_type(8)))  float    v8f;

#define HH 384
#define WW 384
#define BB 4

// ---------------------------------------------------------------------------
// Weight repack: OIHW f32 -> per-lane WMMA B-fragment order, f16.
// dst[((kc*NCT + ct)*32 + lane)*16 + i]
//   kc = tap*(CINP/32) + cc ; lane<16 -> kb=0 ; lane>=16 -> kb=16
//   B[k][n] = W[co=ct*16+(lane&15)][ci=cc*32+kb+i][tap]  (zero-padded)
// ---------------------------------------------------------------------------
__global__ __launch_bounds__(256)
void repack_weights(const float* __restrict__ w, _Float16* __restrict__ dst,
                    int Cin, int Cout, int CINP, int NCT, int total)
{
    int idx = blockIdx.x * blockDim.x + threadIdx.x;
    if (idx >= total) return;
    int i    = idx & 15;
    int lane = (idx >> 4) & 31;
    int ct   = (idx >> 9) % NCT;
    int kc   = idx / (512 * NCT);
    int NCC  = CINP / 32;
    int tap  = kc / NCC;
    int cc   = kc % NCC;
    int kb   = (lane < 16) ? 0 : 16;
    int n    = lane & 15;
    int ci   = cc * 32 + kb + i;
    int co   = ct * 16 + n;
    float v = 0.f;
    if (ci < Cin && co < Cout)
        v = w[(size_t)(co * Cin + ci) * 9 + tap];
    dst[idx] = (_Float16)v;
}

// ---------------------------------------------------------------------------
// 3x3 SAME conv, implicit GEMM via v_wmma_f32_16x16x32_f16.
// Block: 256 threads (8 waves); one row h, 128 pixels [w0,w0+128).
// SRCF16=1: src is f16 NHWC (CINP channels), staged into LDS with
//           global_load_async_to_lds_b128 (ASYNCcnt) for the interior span.
// SRCF16=0: src is f32 NCHW (the network input x), gathered + converted.
// OUTF16=1: dst is f16 NHWC with channel stride COSTR (zero-padded if
//           COUT<COSTR), streamed out via an LDS out-tile as b128 stores.
// OUTF16=0: dst is f32 NCHW (final mask).
// ACT: 0 none, 1 leaky-relu(0.2), 2 sigmoid.
// ---------------------------------------------------------------------------
template<int CINP, int COUT, int COSTR, int NCT, int ACT, int SRCF16, int OUTF16, int CIN>
__global__ __launch_bounds__(256)
void conv3x3_wmma(const void* __restrict__ srcv, const _Float16* __restrict__ packB,
                  const float* __restrict__ bias, void* __restrict__ dstv)
{
    constexpr int TLW = 130;               // 128 + 2 halo
    __shared__ __align__(16) _Float16 lds[3 * TLW * CINP];
    constexpr int OTSZ = OUTF16 ? 128 * COSTR : 8;
    __shared__ __align__(16) _Float16 olds[OTSZ];

    const int w0  = blockIdx.x * 128;
    const int h   = blockIdx.y;
    const int n   = blockIdx.z;
    const int tid = threadIdx.x;

    // zero-init out-tile padding channels (if any) before the staging barrier
    if (OUTF16 && COUT < COSTR) {
        for (int k = tid; k < 128 * COSTR / 8; k += 256)
            ((uint4*)olds)[k] = make_uint4(0u, 0u, 0u, 0u);
    }

    if (SRCF16) {
        const _Float16* s16 = (const _Float16*)srcv;
        bool used_async = false;
        for (int dy = 0; dy < 3; ++dy) {
            const int hh = h + dy - 1;
            _Float16* lrow = &lds[dy * TLW * CINP];
            if (hh < 0 || hh >= HH) {
                for (int k = tid; k < TLW * CINP / 8; k += 256)
                    ((uint4*)lrow)[k] = make_uint4(0u, 0u, 0u, 0u);
            } else {
                const _Float16* g = s16 + ((size_t)(n * HH + hh) * WW + w0) * CINP;
                // interior span: 128 pixels * CINP ch, contiguous in NHWC
                constexpr int NCHUNK = 128 * CINP / 8;     // b128 chunks
                for (int k = tid; k < NCHUNK; k += 256) {
                    unsigned           ldsa = (unsigned)(unsigned long long)&lrow[CINP + k * 8];
                    unsigned long long ga   = (unsigned long long)(g + k * 8);
                    asm volatile("global_load_async_to_lds_b128 %0, %1, off"
                                 :: "v"(ldsa), "v"(ga) : "memory");
                }
                used_async = true;
                // halo columns: pixel w0-1 (wp=0) and w0+128 (wp=129)
                for (int k = tid; k < 2 * CINP; k += 256) {
                    int side = k / CINP, ci = k % CINP;
                    int ww = (side == 0) ? (w0 - 1) : (w0 + 128);
                    _Float16 v = (_Float16)0.f;
                    if (ww >= 0 && ww < WW)
                        v = s16[((size_t)(n * HH + hh) * WW + ww) * CINP + ci];
                    lrow[(side == 0 ? 0 : 129) * CINP + ci] = v;
                }
            }
        }
        if (used_async)
            asm volatile("s_wait_asynccnt 0" ::: "memory");
    } else {
        const float* s32 = (const float*)srcv;
        for (int idx = tid; idx < 3 * TLW * CINP; idx += 256) {
            int wp = idx % TLW;
            int ci = (idx / TLW) % CINP;
            int dy = idx / (TLW * CINP);
            int hh = h + dy - 1;
            int ww = w0 + wp - 1;
            float v = 0.f;
            if (ci < CIN && hh >= 0 && hh < HH && ww >= 0 && ww < WW)
                v = s32[(((size_t)n * CIN + ci) * HH + hh) * WW + ww];
            lds[(dy * TLW + wp) * CINP + ci] = (_Float16)v;
        }
    }
    __syncthreads();

    const int lane = tid & 31;
    const int wave = tid >> 5;
    const int mcol = lane & 15;            // A row M / B-D col N
    const int kbA  = (lane < 16) ? 0 : 16; // A K sub-block: contiguous k = 16h + s

    v8f acc[NCT];
#pragma unroll
    for (int t = 0; t < NCT; ++t) { v8f z = {}; acc[t] = z; }

    constexpr int NCC = CINP / 32;
#pragma unroll
    for (int tap = 0; tap < 9; ++tap) {
        const int dy = tap / 3;
        const int kx = tap % 3;
        const int wp = wave * 16 + mcol + kx;
#pragma unroll
        for (int cc = 0; cc < NCC; ++cc) {
            const int kc = tap * NCC + cc;
            v16h a;
            const _Float16* ap = &lds[(dy * TLW + wp) * CINP + cc * 32 + kbA];
            ((uint4*)&a)[0] = *(const uint4*)(ap);
            ((uint4*)&a)[1] = *(const uint4*)(ap + 8);
#pragma unroll
            for (int ct = 0; ct < NCT; ++ct) {
                v16h b;
                const _Float16* bp = packB + ((size_t)(kc * NCT + ct) * 32 + lane) * 16;
                ((uint4*)&b)[0] = *(const uint4*)(bp);
                ((uint4*)&b)[1] = *(const uint4*)(bp + 8);
                acc[ct] = __builtin_amdgcn_wmma_f32_16x16x32_f16(
                    false, a, false, b, (short)0, acc[ct], false, false);
            }
        }
    }

    const int mb = (lane < 16) ? 0 : 8;    // D rows = consecutive pixels
    if (OUTF16) {
#pragma unroll
        for (int ct = 0; ct < NCT; ++ct) {
            const int co = ct * 16 + mcol;
            if (co < COUT) {
                const float bv = bias[co];
#pragma unroll
                for (int v = 0; v < 8; ++v) {
                    float x = acc[ct][v] + bv;
                    if (ACT == 1)      x = (x >= 0.f) ? x : 0.2f * x;
                    else if (ACT == 2) x = 1.f / (1.f + __expf(-x));
                    olds[(wave * 16 + mb + v) * COSTR + co] = (_Float16)x;
                }
            }
        }
        __syncthreads();
        _Float16* gb = (_Float16*)dstv + ((size_t)(n * HH + h) * WW + w0) * COSTR;
        for (int k = tid; k < 128 * COSTR / 8; k += 256)
            ((uint4*)gb)[k] = ((const uint4*)olds)[k];
    } else {
        float* dst = (float*)dstv;
#pragma unroll
        for (int ct = 0; ct < NCT; ++ct) {
            const int co = ct * 16 + mcol;
            if (co < COUT) {
                const float bv = bias[co];
                float vv[8];
#pragma unroll
                for (int v = 0; v < 8; ++v) {
                    float x = acc[ct][v] + bv;
                    if (ACT == 1)      x = (x >= 0.f) ? x : 0.2f * x;
                    else if (ACT == 2) x = 1.f / (1.f + __expf(-x));
                    vv[v] = x;
                }
                float* op = dst + (((size_t)n * COUT + co) * HH + h) * WW
                                + w0 + wave * 16 + mb;
                *(float4*)(op)     = make_float4(vv[0], vv[1], vv[2], vv[3]);
                *(float4*)(op + 4) = make_float4(vv[4], vv[5], vv[6], vv[7]);
            }
        }
    }
}

// ---------------------------------------------------------------------------
// IRNN directional scan (f32 recurrence) fused with attention multiply.
// src: f16 NHWC 32ch ; att: f16 NHWC 4ch ; cat: f16 NHWC 128ch slice attc*32.
// out[0] = x[0] raw; out[t] = relu(x[t] + alpha*out[t-1] + b).
// ---------------------------------------------------------------------------
template<int DIRV>  // +1: down, -1: up
__global__ __launch_bounds__(256)
void scan_vert_mul(const _Float16* __restrict__ src, const _Float16* __restrict__ att,
                   const float* __restrict__ wch, const float* __restrict__ bch,
                   _Float16* __restrict__ cat, int attc)
{
    int idx = blockIdx.x * blockDim.x + threadIdx.x;
    if (idx >= BB * 32 * WW) return;
    int c = idx & 31;
    int w = (idx >> 5) % WW;
    int n = idx / (32 * WW);
    const float alpha = wch[c];
    const float bb    = bch[c];
    const int h0 = (DIRV > 0) ? 0 : HH - 1;
    long long p = (long long)(n * HH + h0) * WW + w;
    const long long pstep = (long long)DIRV * WW;
    float hp = (float)src[p * 32 + c];
    cat[p * 128 + attc * 32 + c] = (_Float16)(hp * (float)att[p * 4 + attc]);
    for (int t = 1; t < HH; ++t) {
        p += pstep;
        float v = (float)src[p * 32 + c] + alpha * hp + bb;
        v = fmaxf(v, 0.f);
        cat[p * 128 + attc * 32 + c] = (_Float16)(v * (float)att[p * 4 + attc]);
        hp = v;
    }
}

template<int DIRV>  // +1: right, -1: left
__global__ __launch_bounds__(256)
void scan_horz_mul(const _Float16* __restrict__ src, const _Float16* __restrict__ att,
                   const float* __restrict__ wch, const float* __restrict__ bch,
                   _Float16* __restrict__ cat, int attc)
{
    int idx = blockIdx.x * blockDim.x + threadIdx.x;
    if (idx >= BB * 32 * HH) return;
    int c = idx & 31;
    int h = (idx >> 5) % HH;
    int n = idx / (32 * HH);
    const float alpha = wch[c];
    const float bb    = bch[c];
    const int w0 = (DIRV > 0) ? 0 : WW - 1;
    long long p = (long long)(n * HH + h) * WW + w0;
    float hp = (float)src[p * 32 + c];
    cat[p * 128 + attc * 32 + c] = (_Float16)(hp * (float)att[p * 4 + attc]);
    for (int t = 1; t < WW; ++t) {
        p += DIRV;
        float v = (float)src[p * 32 + c] + alpha * hp + bb;
        v = fmaxf(v, 0.f);
        cat[p * 128 + attc * 32 + c] = (_Float16)(v * (float)att[p * 4 + attc]);
        hp = v;
    }
}

// ---------------------------------------------------------------------------
extern "C" void kernel_launch(void* const* d_in, const int* in_sizes, int n_in,
                              void* d_out, int out_size, void* d_ws, size_t ws_size,
                              hipStream_t stream)
{
    (void)in_sizes; (void)n_in; (void)out_size; (void)ws_size;

    const float* x          = (const float*)d_in[0];
    const float* att_w1     = (const float*)d_in[1];
    const float* att_b1     = (const float*)d_in[2];
    const float* att_w2     = (const float*)d_in[3];
    const float* att_b2     = (const float*)d_in[4];
    const float* att_w3     = (const float*)d_in[5];
    const float* att_b3     = (const float*)d_in[6];
    const float* conv_in_w  = (const float*)d_in[7];
    const float* conv_in_b  = (const float*)d_in[8];
    const float* conv2_w    = (const float*)d_in[9];
    const float* conv2_b    = (const float*)d_in[10];
    const float* conv3_w    = (const float*)d_in[11];
    const float* conv3_b    = (const float*)d_in[12];
    const float* conv_out_w = (const float*)d_in[13];
    const float* conv_out_b = (const float*)d_in[14];
    const float* r1p[8]; const float* r2p[8];
    for (int i = 0; i < 8; ++i) { r1p[i] = (const float*)d_in[15 + i]; r2p[i] = (const float*)d_in[23 + i]; }

    const size_t NHW = (size_t)BB * HH * WW;       // 589824 pixels
    _Float16* a1   = (_Float16*)d_ws;              // 32ch padded NHWC (16 real)
    _Float16* a2   = a1  + 32 * NHW;               // 32ch padded NHWC
    _Float16* att  = a2  + 32 * NHW;               // 4ch NHWC
    _Float16* cat  = att + 4  * NHW;               // 128ch NHWC
    _Float16* outb = a1;                           // reuse a1 (32ch) after att done
    _Float16* pk    = cat + (size_t)128 * NHW;
    _Float16* pk_a1 = pk;                          // 9*1*1*512 = 4608
    _Float16* pk_a2 = pk_a1 + 4608;
    _Float16* pk_a3 = pk_a2 + 4608;
    _Float16* pk_in = pk_a3 + 4608;                // 9*1*2*512 = 9216
    _Float16* pk_c2 = pk_in + 9216;                // 9*4*2*512 = 36864
    _Float16* pk_c3 = pk_c2 + 36864;
    _Float16* pk_co = pk_c3 + 36864;

    auto rp = [&](const float* w, _Float16* dst, int Cin, int Cout, int CINP, int NCT) {
        int total = 9 * (CINP / 32) * NCT * 512;
        repack_weights<<<(total + 255) / 256, 256, 0, stream>>>(w, dst, Cin, Cout, CINP, NCT, total);
    };
    rp(att_w1,     pk_a1,  32, 16,  32, 1);
    rp(att_w2,     pk_a2,  16, 16,  32, 1);
    rp(att_w3,     pk_a3,  16,  4,  32, 1);
    rp(conv_in_w,  pk_in,  32, 32,  32, 2);
    rp(conv2_w,    pk_c2, 128, 32, 128, 2);
    rp(conv3_w,    pk_c3, 128, 32, 128, 2);
    rp(conv_out_w, pk_co,  32, 32,  32, 2);

    const dim3 cg(WW / 128, HH, BB), cb(256);
    const int  nscan = BB * 32 * WW;
    const dim3 sg((nscan + 255) / 256), sb(256);

    // attention branch                 CINP,COUT,COSTR,NCT,ACT,SRCF16,OUTF16,CIN
    conv3x3_wmma<32, 16, 32, 1, 1, 0, 1, 32><<<cg, cb, 0, stream>>>(x,  pk_a1, att_b1, a1);
    conv3x3_wmma<32, 16, 32, 1, 1, 1, 1, 32><<<cg, cb, 0, stream>>>(a1, pk_a2, att_b2, a2);
    conv3x3_wmma<32,  4,  4, 1, 2, 1, 1, 32><<<cg, cb, 0, stream>>>(a2, pk_a3, att_b3, att);

    // main branch
    conv3x3_wmma<32, 32, 32, 2, 0, 0, 1, 32><<<cg, cb, 0, stream>>>(x, pk_in, conv_in_b, outb);

    // IRNN round 1: concat order [u, r, d, l]; attention channels 0,1,2,3
    scan_vert_mul<-1><<<sg, sb, 0, stream>>>(outb, att, r1p[0], r1p[1], cat, 0);
    scan_horz_mul<+1><<<sg, sb, 0, stream>>>(outb, att, r1p[2], r1p[3], cat, 1);
    scan_vert_mul<+1><<<sg, sb, 0, stream>>>(outb, att, r1p[4], r1p[5], cat, 2);
    scan_horz_mul<-1><<<sg, sb, 0, stream>>>(outb, att, r1p[6], r1p[7], cat, 3);

    conv3x3_wmma<128, 32, 32, 2, 0, 1, 1, 128><<<cg, cb, 0, stream>>>(cat, pk_c2, conv2_b, outb);

    // IRNN round 2
    scan_vert_mul<-1><<<sg, sb, 0, stream>>>(outb, att, r2p[0], r2p[1], cat, 0);
    scan_horz_mul<+1><<<sg, sb, 0, stream>>>(outb, att, r2p[2], r2p[3], cat, 1);
    scan_vert_mul<+1><<<sg, sb, 0, stream>>>(outb, att, r2p[4], r2p[5], cat, 2);
    scan_horz_mul<-1><<<sg, sb, 0, stream>>>(outb, att, r2p[6], r2p[7], cat, 3);

    conv3x3_wmma<128, 32, 32, 2, 1, 1, 1, 128><<<cg, cb, 0, stream>>>(cat, pk_c3, conv3_b, outb);

    // final mask: f32 NCHW to d_out
    conv3x3_wmma<32, 32, 32, 2, 2, 1, 0, 32><<<cg, cb, 0, stream>>>(outb, pk_co, conv_out_b, (float*)d_out);
}
